// EdgeLearner_76063870812179
// MI455X (gfx1250) — matrix operand order repaired and checked by reading, not compiled, not run on hardware
//
#include <hip/hip_runtime.h>

// Problem constants (fixed by the reference).
#define BB 16
#define VV 19
#define EE 361
#define LL 256
#define DD 64

// Padded bf16 weight image (in scratch and mirrored in LDS):
//   W1T: [n=64][k stride 160]  k<=128 -> W1[k][n], k==129 -> b1[n], else 0
//        (k=128 is the edge-weight column, k=129 is the bias column -> folded GEMM)
//   W2T: [n=32][k stride 80]   k<64  -> W2[k][n], else 0
#define W1T_STRIDE 160
#define W2T_STRIDE 80
#define W1T_ELEMS (64 * W1T_STRIDE)
#define W2T_ELEMS (32 * W2T_STRIDE)
#define WIMG_ELEMS (W1T_ELEMS + W2T_ELEMS)       // 12800 bf16
#define WIMG_BYTES (WIMG_ELEMS * 2)              // 25600 B
#define HSB_ELEMS (BB * VV * LL * DD)            // 4,980,736 bf16
#define WS_NEEDED ((size_t)WIMG_BYTES + (size_t)HSB_ELEMS * 2)

typedef __attribute__((ext_vector_type(16))) __bf16 v16bf;
typedef __attribute__((ext_vector_type(8)))  __bf16 v8bf;
typedef __attribute__((ext_vector_type(4)))  __bf16 v4bf;
typedef __attribute__((ext_vector_type(8)))  float  v8f;

__device__ __forceinline__ float sigmoidf_(float x) { return 1.0f / (1.0f + __expf(-x)); }
__device__ __forceinline__ float siluf_(float x)    { return x * sigmoidf_(x); }

// B-matrix fragment: 8 bf16 K-values at p and 8 at p+16 for this lane's column.
// 16B-aligned by construction -> two ds_load_b128.
__device__ __forceinline__ v16bf lds_bfrag(const __bf16* p) {
  v8bf lo = *(const v8bf*)p, hi = *(const v8bf*)(p + 16);
  v16bf r;
#pragma unroll
  for (int i = 0; i < 8; ++i) { r[i] = lo[i]; r[8 + i] = hi[i]; }
  return r;
}

__device__ __forceinline__ void pack8f(v16bf& a, int off, float4 f0, float4 f1) {
  a[off + 0] = (__bf16)f0.x; a[off + 1] = (__bf16)f0.y;
  a[off + 2] = (__bf16)f0.z; a[off + 3] = (__bf16)f0.w;
  a[off + 4] = (__bf16)f1.x; a[off + 5] = (__bf16)f1.y;
  a[off + 6] = (__bf16)f1.z; a[off + 7] = (__bf16)f1.w;
}

// ---------------- pre-pass 1: padded/transposed bf16 weight image --------------
__global__ __launch_bounds__(256) void prep_weights_kernel(
    const float* __restrict__ W1, const float* __restrict__ b1,
    const float* __restrict__ W2, __bf16* __restrict__ wimg) {
  const int tid = threadIdx.x;
  for (int i = tid; i < W1T_ELEMS; i += 256) {
    int n = i / W1T_STRIDE, k = i - n * W1T_STRIDE;
    float v = 0.0f;
    if (k <= 128) v = W1[k * 64 + n];
    else if (k == 129) v = b1[n];
    wimg[i] = (__bf16)v;
  }
  __bf16* w2 = wimg + W1T_ELEMS;
  for (int i = tid; i < W2T_ELEMS; i += 256) {
    int n = i / W2T_STRIDE, k = i - n * W2T_STRIDE;
    float v = (k < 64) ? W2[k * 32 + n] : 0.0f;
    w2[i] = (__bf16)v;
  }
}

// ---------------- pre-pass 2: hidden_states f32 -> bf16 (streaming) ------------
__global__ __launch_bounds__(256) void prep_hs_kernel(const float* __restrict__ hs,
                                                      __bf16* __restrict__ hsb) {
  const int i = (blockIdx.x * 256 + threadIdx.x) * 4;
  float4 f = *(const float4*)(hs + i);
  v4bf o;
  o[0] = (__bf16)f.x; o[1] = (__bf16)f.y; o[2] = (__bf16)f.z; o[3] = (__bf16)f.w;
  *(v4bf*)(hsb + i) = o;
}

// ---------------- main kernel --------------------------------------------------
template <bool PREP>
__global__ __launch_bounds__(256) void edge_mlp_wmma_kernel(
    const float*  __restrict__ hs,    // (B*V, L, D) f32 (fallback gather)
    const __bf16* __restrict__ hsb,   // same, bf16 (PREP gather)
    const __bf16* __restrict__ wimg,  // prepped weight image (PREP)
    const float* __restrict__ ew,
    const float* __restrict__ W1, const float* __restrict__ b1,
    const float* __restrict__ W2, const float* __restrict__ b2,
    const float* __restrict__ W3, const float* __restrict__ b3,
    const float* __restrict__ skipp, const int* __restrict__ eidx,
    float* __restrict__ out)
{
  __shared__ __align__(32) __bf16 sWT[WIMG_ELEMS];   // 25.6 KB
  __shared__ __align__(32) __bf16 sH1[8 * 16 * 72];  // 18.4 KB per-wave h1 tiles

  const int tid = threadIdx.x;

  // ---- stage weight image into LDS ----
  if constexpr (PREP) {
    const uint4* src = (const uint4*)wimg;
    uint4* dst = (uint4*)sWT;
    for (int i = tid; i < WIMG_BYTES / 16; i += 256) dst[i] = src[i];
  } else {
    for (int i = tid; i < W1T_ELEMS; i += 256) {
      int n = i / W1T_STRIDE, k = i - n * W1T_STRIDE;
      float v = 0.0f;
      if (k <= 128) v = W1[k * 64 + n];
      else if (k == 129) v = b1[n];
      sWT[i] = (__bf16)v;
    }
    for (int i = tid; i < W2T_ELEMS; i += 256) {
      int n = i / W2T_STRIDE, k = i - n * W2T_STRIDE;
      sWT[W1T_ELEMS + i] = (__bf16)((k < 64) ? W2[k * 32 + n] : 0.0f);
    }
  }
  __syncthreads();
  const __bf16* sW1T = sWT;
  const __bf16* sW2T = sWT + W1T_ELEMS;

  const int wave = tid >> 5;
  const int lane = tid & 31;
  const int m    = lane & 15;   // A-row / B,D-column within the 16-wide tile
  const int half = lane >> 4;   // K-half (A/B layouts), row-half (C/D layout)
  const int wg   = blockIdx.x * 8 + wave;
  const int be   = wg >> 4;            // edge id in (B*E)
  const int l0   = (wg & 15) << 4;     // this wave's 16 timesteps
  const int b    = be / EE;
  const int l    = l0 + m;

  const float b3v  = b3[0];
  const float skip = skipp[0];

  // ---- per-lane gather indices (this lane's A-row = timestep l0+m) ----
  const int ii = be * LL + l;
  unsigned sV = ((unsigned)eidx[ii]) % VV;
  unsigned tV = ((unsigned)eidx[BB * EE * LL + ii]) % VV;
  const size_t srcOff = ((size_t)(b * VV + sV) * LL + l) * DD;
  const size_t tgtOff = ((size_t)(b * VV + tV) * LL + l) * DD;

  // Edge weights for this lane's accumulator rows (C/D rows half*8 .. half*8+7).
  const float* ewRow = ew + (size_t)be * LL + l0 + half * 8;
  float4 e0 = *(const float4*)(ewRow);
  float4 e1 = *(const float4*)(ewRow + 4);
  float ewv[8] = {e0.x, e0.y, e0.z, e0.w, e1.x, e1.y, e1.z, e1.w};
  // This lane's own-row edge weight (feeds the folded k=128 column).
  float ewl = ew[(size_t)be * LL + l];

  // ================ GEMM1: x(16x130) @ W1ext(130x64), K tiles {0..4} ==========
  // Tiles 0-1 = src features, 2-3 = tgt features, tile 4 = [ew, 1, 0...] fold.
  v16bf a1[5];
  if constexpr (PREP) {
    const __bf16* srcB = hsb + srcOff;
    const __bf16* tgtB = hsb + tgtOff;
#pragma unroll
    for (int ks = 0; ks < 4; ++ks) {
      const __bf16* base = (ks < 2) ? srcB : tgtB;
      int kA = ((ks & 1) << 5) + half * 8;     // elems 0-7: K0+half*8, 8-15: +16
      v8bf lo = *(const v8bf*)(base + kA);
      v8bf hi = *(const v8bf*)(base + kA + 16);
#pragma unroll
      for (int i = 0; i < 8; ++i) { a1[ks][i] = lo[i]; a1[ks][8 + i] = hi[i]; }
    }
  } else {
    const float* srcF = hs + srcOff;
    const float* tgtF = hs + tgtOff;
#pragma unroll
    for (int ks = 0; ks < 4; ++ks) {
      const float* base = (ks < 2) ? srcF : tgtF;
      int kA = ((ks & 1) << 5) + half * 8;
      float4 f0 = *(const float4*)(base + kA);
      float4 f1 = *(const float4*)(base + kA + 4);
      float4 f2 = *(const float4*)(base + kA + 16);
      float4 f3 = *(const float4*)(base + kA + 20);
      pack8f(a1[ks], 0, f0, f1);
      pack8f(a1[ks], 8, f2, f3);
    }
  }
  // Tile 4 (k=128..159): only half==0 lanes carry k=128 (ew) and k=129 (1.0).
#pragma unroll
  for (int i = 0; i < 16; ++i) a1[4][i] = (__bf16)0.0f;
  a1[4][0] = (half == 0) ? (__bf16)ewl : (__bf16)0.0f;
  a1[4][1] = (half == 0) ? (__bf16)1.0f : (__bf16)0.0f;

  __bf16* sH1w = sH1 + wave * (16 * 72);
#pragma unroll
  for (int nt = 0; nt < 4; ++nt) {
    int n = nt * 16 + m;                 // this lane's output column
    v8f acc = {};                        // bias folded into K -> start from zero
#pragma unroll
    for (int ks = 0; ks < 5; ++ks) {
      v16bf bf = lds_bfrag(&sW1T[n * W1T_STRIDE + ks * 32 + half * 8]);
      acc = __builtin_amdgcn_wmma_f32_16x16x32_bf16(false, a1[ks], false, bf,
                                                    (short)0, acc, false, false);
    }
    // silu, stash to LDS for the D->A layout transpose (LDS in-order per wave).
#pragma unroll
    for (int v = 0; v < 8; ++v)
      sH1w[(v + 8 * half) * 72 + n] = (__bf16)siluf_(acc[v]);
  }

  // ================ GEMM2: h1(16x64) @ W2(64x32) ==============================
  v16bf a2[2];
#pragma unroll
  for (int ks = 0; ks < 2; ++ks) {
    int kA = ks * 32 + half * 8;
    v8bf lo = *(const v8bf*)&sH1w[m * 72 + kA];
    v8bf hi = *(const v8bf*)&sH1w[m * 72 + kA + 16];
#pragma unroll
    for (int i = 0; i < 8; ++i) { a2[ks][i] = lo[i]; a2[ks][8 + i] = hi[i]; }
  }

  float h2[2][8];
#pragma unroll
  for (int nt = 0; nt < 2; ++nt) {
    int n = nt * 16 + m;
    float bias = b2[n];                  // L1/L2-resident scalar
    v8f acc;
#pragma unroll
    for (int v = 0; v < 8; ++v) acc[v] = bias;
#pragma unroll
    for (int ks = 0; ks < 2; ++ks) {
      v16bf bf = lds_bfrag(&sW2T[n * W2T_STRIDE + ks * 32 + half * 8])
      acc = __builtin_amdgcn_wmma_f32_16x16x32_bf16(false, a2[ks], false, bf,
                                                    (short)0, acc, false, false);
    }
#pragma unroll
    for (int v = 0; v < 8; ++v) h2[nt][v] = siluf_(acc[v]);
  }

  // ========= GEMM3 (N=1): VALU dot with W3 + 16-lane xor reduction ============
  float w3a = W3[m];
  float w3b = W3[16 + m];
  float c[8];
#pragma unroll
  for (int v = 0; v < 8; ++v) c[v] = h2[0][v] * w3a + h2[1][v] * w3b;
#pragma unroll
  for (int v = 0; v < 8; ++v) {
    c[v] += __shfl_xor(c[v], 1, 32);
    c[v] += __shfl_xor(c[v], 2, 32);
    c[v] += __shfl_xor(c[v], 4, 32);
    c[v] += __shfl_xor(c[v], 8, 32);    // masks stay inside the 16-lane half
  }
  if (m == 0) {
    float* o = out + (size_t)be * LL + l0 + 8 * half;
#pragma unroll
    for (int v = 0; v < 8; ++v) {
      float wfin = sigmoidf_(c[v] + b3v);
      o[v] = skip * ewv[v] + (1.0f - skip) * wfin;
    }
  }
}

extern "C" void kernel_launch(void* const* d_in, const int* in_sizes, int n_in,
                              void* d_out, int out_size, void* d_ws, size_t ws_size,
                              hipStream_t stream) {
  const float* hs  = (const float*)d_in[0];
  const float* ew  = (const float*)d_in[1];
  const float* W1  = (const float*)d_in[2];
  const float* b1  = (const float*)d_in[3];
  const float* W2  = (const float*)d_in[4];
  const float* b2  = (const float*)d_in[5];
  const float* W3  = (const float*)d_in[6];
  const float* b3  = (const float*)d_in[7];
  const float* sk  = (const float*)d_in[8];
  const int*   ei  = (const int*)d_in[9];
  float* out = (float*)d_out;

  // 8 waves/block * 16 rows/wave = 128 rows/block; N = B*E*L -> 11552 blocks exact.
  const int nblocks = (BB * EE * LL) / 128;

  if (ws_size >= WS_NEEDED) {
    __bf16* wimg = (__bf16*)d_ws;
    __bf16* hsb  = (__bf16*)((char*)d_ws + WIMG_BYTES);
    prep_weights_kernel<<<1, 256, 0, stream>>>(W1, b1, W2, wimg);
    prep_hs_kernel<<<HSB_ELEMS / 1024, 256, 0, stream>>>(hs, hsb);
    edge_mlp_wmma_kernel<true><<<nblocks, 256, 0, stream>>>(
        hs, hsb, wimg, ew, W1, b1, W2, b2, W3, b3, sk, ei, out);
  } else {
    edge_mlp_wmma_kernel<false><<<nblocks, 256, 0, stream>>>(
        hs, nullptr, nullptr, ew, W1, b1, W2, b2, W3, b3, sk, ei, out);
  }
}
